// StateSpaceLayer_31095563223441
// MI455X (gfx1250) — matrix operand order repaired and checked by reading, not compiled, not run on hardware
//
#include <hip/hip_runtime.h>
#include <hip/hip_bf16.h>

// ---------------------------------------------------------------------------
// StateSpaceLayer (Mamba-like) for MI455X / gfx1250.
// GEMMs in bf16 via v_wmma_f32_16x16x32_bf16 (f32 accumulate), register-blocked
// 32x64 per wave (8 wmma / k-step), ping-pong software-pipelined k-loop
// (unroll by 64, no rotation copies, no spills with launch_bounds(256,1)).
// Conv / scan / gating epilogues in fp32 VALU. wave32 everywhere.
// ---------------------------------------------------------------------------

typedef __attribute__((ext_vector_type(16))) __bf16 v16bf;
typedef __attribute__((ext_vector_type(8)))  __bf16 v8bf;
typedef __attribute__((ext_vector_type(8)))  float  v8f;

// Problem constants (match reference).
#define BB  2
#define LL  2048
#define DD  768
#define DII 1536
#define DSS 16
#define KCV 4
#define MLT (BB * LL)   // 4096 token rows

// ---------------------------------------------------------------------------
// fp32 -> bf16 elementwise convert
// ---------------------------------------------------------------------------
__global__ void cvt_bf16_kernel(const float* __restrict__ in,
                                __bf16* __restrict__ out, long n) {
  long i = (long)blockIdx.x * blockDim.x + threadIdx.x;
  if (i < n) out[i] = (__bf16)in[i];
}

// ---------------------------------------------------------------------------
// fp32 (R x C, row-major) -> bf16 transposed (C x R, row-major)
// out[c*R + r] = in[r*C + c]
// ---------------------------------------------------------------------------
__global__ void transpose_cvt_kernel(const float* __restrict__ in,
                                     __bf16* __restrict__ out, int R, int C) {
  long i = (long)blockIdx.x * blockDim.x + threadIdx.x;
  long total = (long)R * C;
  if (i >= total) return;
  int c = (int)(i / R);
  int r = (int)(i % R);
  out[i] = (__bf16)in[(long)r * C + c];
}

// ---------------------------------------------------------------------------
// Fragment loader: 16 bf16 per lane as two 16B (b128) loads.
//   A frag: second half at +16 elements (ISA 16x32 bf16 A layout)
//   B frag: not loaded here; loadB gathers 16 k values of one N column
// ---------------------------------------------------------------------------
__device__ __forceinline__ v16bf ldfrag(const __bf16* p, int second_off) {
  v8bf lo = *(const v8bf*)p;
  v8bf hi = *(const v8bf*)(p + second_off);
  v16bf r;
#pragma unroll
  for (int i = 0; i < 8; ++i) { r[i] = lo[i]; r[i + 8] = hi[i]; }
  return r;
}

__device__ __forceinline__ v16bf zfrag() {
  v16bf z;
#pragma unroll
  for (int i = 0; i < 16; ++i) z[i] = (__bf16)0.0f;
  return z;
}

// ---------------------------------------------------------------------------
// C[M,N] (f32) = A[M,K] (bf16, row-major) * Bt[K,N] (bf16, row-major).
//
// Block = 256 threads = 8 waves (2 x 4). Wave tile = 32(M) x 64(N):
//   2 A fragments x 4 B fragments -> 8 f32 accumulators, 8 wmma per k-step.
// Block tile = 64(M) x 256(N). Requires K % 64 == 0, M % 64 == 0.
// Ping-pong pipelining: while wmma consumes fragment set 0, set 1 loads,
// and vice versa -- no inter-buffer copies, no spills (launch_bounds(256,1)).
// NGUARD=true adds wave-uniform per-16-col tile guards (for N=32 case).
// ---------------------------------------------------------------------------
template <bool NGUARD>
__global__ __launch_bounds__(256, 1)
void gemm_bf16_wmma_kernel(const __bf16* __restrict__ A,
                           const __bf16* __restrict__ Bt,
                           float* __restrict__ C,
                           int M, int N, int K) {
  const int lane = threadIdx.x & 31;
  const int wave = threadIdx.x >> 5;
  const int wm   = wave >> 2;                 // 0..1
  const int wn   = wave & 3;                  // 0..3
  const int m0   = blockIdx.y * 64 + wm * 32; // always < M (M % 64 == 0)
  const int n0   = blockIdx.x * 256 + wn * 64;
  if (NGUARD && n0 >= N) return;              // wave-uniform

  const int l15 = lane & 15;
  const int h   = lane >> 4;

  // A row pointers for the two m-tiles (m0, m0+16), K-offset h*8 per layout.
  const __bf16* aptr0 = A + (long)(m0 + l15) * K + h * 8;
  const __bf16* aptr1 = aptr0 + (long)16 * K;
  // B base: lane = column n0 + 16*j + l15, K-rows h*8 (+16) per layout.
  const __bf16* bbase = Bt + (long)(h * 8) * N + n0 + l15;

  bool nv[4];
#pragma unroll
  for (int j = 0; j < 4; ++j)
    nv[j] = !NGUARD || (n0 + 16 * j) < N;

  v8f acc[2][4];
#pragma unroll
  for (int i = 0; i < 2; ++i)
#pragma unroll
    for (int j = 0; j < 4; ++j) {
      v8f z = {};
      acc[i][j] = z;
    }

  auto loadA = [&](int kt, v16bf av[2]) {
    av[0] = ldfrag(aptr0 + kt, 16);
    av[1] = ldfrag(aptr1 + kt, 16);
  };
  auto loadB = [&](int kt, v16bf bv[4]) {
    const __bf16* bp = bbase + (long)kt * N;
#pragma unroll
    for (int j = 0; j < 4; ++j) {
      if (nv[j]) {
        v16bf t;
#pragma unroll
        for (int s = 0; s < 8; ++s) {
          t[s] = bp[(long)s * N + 16 * j];
          t[s + 8] = bp[(long)(16 + s) * N + 16 * j];
        }
        bv[j] = t;
      } else {
        bv[j] = zfrag();
      }
    }
  };
  auto wmma8 = [&](v16bf av[2], v16bf bv[4]) {
#pragma unroll
    for (int i = 0; i < 2; ++i)
#pragma unroll
      for (int j = 0; j < 4; ++j)
        acc[i][j] = __builtin_amdgcn_wmma_f32_16x16x32_bf16(
            false, av[i], false, bv[j], (short)0, acc[i][j], false, false);
  };

  // Ping-pong buffers; K % 64 == 0 so kt+32 is always in range inside loop.
  v16bf a0[2], b0[4], a1[2], b1[4];
  loadA(0, a0);
  loadB(0, b0);

  for (int kt = 0; kt < K; kt += 64) {
    loadA(kt + 32, a1);          // loads overlap wmma8(a0,b0)
    loadB(kt + 32, b1);
    if (kt + 64 < K) {
      __builtin_prefetch(aptr0 + kt + 64, 0, 0);
      __builtin_prefetch(bbase + (long)(kt + 64) * N, 0, 0);
    }
    wmma8(a0, b0);
    if (kt + 64 < K) {
      loadA(kt + 64, a0);        // loads overlap wmma8(a1,b1)
      loadB(kt + 64, b0);
    }
    wmma8(a1, b1);
  }

  // ---- store D: element r -> row m0+16i+h*8+r, col n0+16j+l15 ----
#pragma unroll
  for (int i = 0; i < 2; ++i) {
    const int rbase = m0 + 16 * i + h * 8;
#pragma unroll
    for (int j = 0; j < 4; ++j) {
      if (!nv[j]) continue;
      const int col = n0 + 16 * j + l15;
#pragma unroll
      for (int r = 0; r < 8; ++r)
        C[(long)(rbase + r) * N + col] = acc[i][j][r];
    }
  }
}

// ---------------------------------------------------------------------------
// Depthwise causal conv (K=4) + bias + SiLU on xz[..., :DI]; emits bf16 xc.
// xz layout: (B*L, 2*DI) fp32.
// ---------------------------------------------------------------------------
__global__ void conv_silu_kernel(const float* __restrict__ xz,
                                 const float* __restrict__ conv_w,
                                 const float* __restrict__ conv_b,
                                 __bf16* __restrict__ xcc) {
  long idx = (long)blockIdx.x * blockDim.x + threadIdx.x;
  const long total = (long)MLT * DII;
  if (idx >= total) return;
  int  c  = (int)(idx % DII);
  long bl = idx / DII;
  int  l  = (int)(bl % LL);
  long b0 = bl - l;                    // b*LL
  float acc = conv_b[c];
#pragma unroll
  for (int k = 0; k < KCV; ++k) {
    int lp = l - (KCV - 1) + k;
    if (lp >= 0)
      acc += xz[(b0 + lp) * (2 * DII) + c] * conv_w[c * KCV + k];
  }
  float sig = 1.0f / (1.0f + __expf(-acc));
  xcc[idx] = (__bf16)(acc * sig);
}

// ---------------------------------------------------------------------------
// Sequential selective scan. grid = (DI/256, B), block = 256 threads; each
// thread owns one channel c and h[16] states in registers. A_t/B_t (32 f32)
// staged through LDS per timestep. softplus(dt) + SiLU(z) gating fused.
// ---------------------------------------------------------------------------
__global__ void scan_kernel(const float* __restrict__ abc,   // (B*L, 32)
                            const float* __restrict__ dtpre, // (B*L, DI)
                            const float* __restrict__ b_dt,  // (DI)
                            const __bf16* __restrict__ xcc,  // (B*L, DI)
                            const float* __restrict__ xz,    // (B*L, 2*DI)
                            __bf16* __restrict__ ygbf) {     // (B*L, DI)
  const int b = blockIdx.y;
  const int c = blockIdx.x * blockDim.x + threadIdx.x;
  __shared__ float sA[DSS];
  __shared__ float sB[DSS];

  float h[DSS];
#pragma unroll
  for (int s = 0; s < DSS; ++s) h[s] = 0.0f;
  const float bdt = b_dt[c];

  for (int t = 0; t < LL; ++t) {
    const long bl = (long)b * LL + t;
    if (threadIdx.x < 2 * DSS) {
      float v = abc[bl * (2 * DSS) + threadIdx.x];
      if (threadIdx.x < DSS) sA[threadIdx.x] = v;
      else                   sB[threadIdx.x - DSS] = v;
    }
    __syncthreads();

    float dtv = dtpre[bl * DII + c] + bdt;
    dtv = (dtv > 20.0f) ? dtv : log1pf(__expf(dtv));   // softplus
    float xv = (float)xcc[bl * DII + c];

    float acc = 0.0f;
#pragma unroll
    for (int s = 0; s < DSS; ++s) {
      float a = __expf(sA[s] * dtv);
      h[s] = a * h[s] + sB[s] * xv;
      acc += h[s] * sB[s];
    }

    float zv = xz[bl * (2 * DII) + DII + c];
    float g  = zv / (1.0f + __expf(-zv));              // SiLU(z)
    ygbf[bl * DII + c] = (__bf16)(acc * g);
    __syncthreads();
  }
}

// ---------------------------------------------------------------------------
// Launch
// ---------------------------------------------------------------------------
extern "C" void kernel_launch(void* const* d_in, const int* in_sizes, int n_in,
                              void* d_out, int out_size, void* d_ws, size_t ws_size,
                              hipStream_t stream) {
  const float* x      = (const float*)d_in[0];  // (B,L,D)
  const float* W_in   = (const float*)d_in[1];  // (2*DI, D)
  const float* conv_w = (const float*)d_in[2];  // (DI,1,K)
  const float* conv_b = (const float*)d_in[3];  // (DI)
  const float* W_x    = (const float*)d_in[4];  // (2*DS, DI)
  const float* W_dt   = (const float*)d_in[5];  // (DI, DI)
  const float* b_dt   = (const float*)d_in[6];  // (DI)
  const float* W_out  = (const float*)d_in[7];  // (D, DI)
  float* out = (float*)d_out;                   // (B,L,D)

  // ---- workspace carve-up (256B aligned) ----
  char* p = (char*)d_ws;
  auto carve = [&](size_t bytes) -> char* {
    char* r = p;
    p += (bytes + 255) & ~(size_t)255;
    return r;
  };
  __bf16* x_bf   = (__bf16*)carve((size_t)MLT * DD  * 2);          // (B*L, D)
  __bf16* Wint   = (__bf16*)carve((size_t)DD  * 2 * DII * 2);      // (D, 2*DI)
  __bf16* Wdtt   = (__bf16*)carve((size_t)DII * DII * 2);          // (DI, DI)
  __bf16* Wxt    = (__bf16*)carve((size_t)DII * 2 * DSS * 2);      // (DI, 2*DS)
  __bf16* Woutt  = (__bf16*)carve((size_t)DII * DD  * 2);          // (DI, D)
  float*  xz     = (float*) carve((size_t)MLT * 2 * DII * 4);      // (B*L, 2*DI)
  __bf16* xcc    = (__bf16*)carve((size_t)MLT * DII * 2);          // (B*L, DI)
  float*  dtpre  = (float*) carve((size_t)MLT * DII * 4);          // (B*L, DI)
  float*  abc    = (float*) carve((size_t)MLT * 2 * DSS * 4);      // (B*L, 32)
  __bf16* ygbf   = (__bf16*)carve((size_t)MLT * DII * 2);          // (B*L, DI)

  const int T = 256;
  auto blocks = [](long n, int t) { return (int)((n + t - 1) / t); };

  // 1) convert x -> bf16
  cvt_bf16_kernel<<<blocks((long)MLT * DD, T), T, 0, stream>>>(
      x, x_bf, (long)MLT * DD);

  // 2) transpose+convert weights (out[c*R+r] = in[r*C+c])
  transpose_cvt_kernel<<<blocks((long)(2 * DII) * DD, T), T, 0, stream>>>(
      W_in, Wint, 2 * DII, DD);                      // -> (D, 2*DI)
  transpose_cvt_kernel<<<blocks((long)DII * DII, T), T, 0, stream>>>(
      W_dt, Wdtt, DII, DII);                         // -> (DI, DI)
  transpose_cvt_kernel<<<blocks((long)(2 * DSS) * DII, T), T, 0, stream>>>(
      W_x, Wxt, 2 * DSS, DII);                       // -> (DI, 2*DS)
  transpose_cvt_kernel<<<blocks((long)DD * DII, T), T, 0, stream>>>(
      W_out, Woutt, DD, DII);                        // -> (DI, D)

  // 3) xz = x @ W_in^T : M=4096, N=3072, K=768
  {
    dim3 g((2 * DII + 255) / 256, (MLT + 63) / 64);
    gemm_bf16_wmma_kernel<false><<<g, 256, 0, stream>>>(x_bf, Wint, xz,
                                                        MLT, 2 * DII, DD);
  }

  // 4) depthwise causal conv + bias + SiLU -> xcc (bf16)
  conv_silu_kernel<<<blocks((long)MLT * DII, T), T, 0, stream>>>(
      xz, conv_w, conv_b, xcc);

  // 5) dt_pre = xc @ W_dt^T : M=4096, N=1536, K=1536
  {
    dim3 g((DII + 255) / 256, (MLT + 63) / 64);
    gemm_bf16_wmma_kernel<false><<<g, 256, 0, stream>>>(xcc, Wdtt, dtpre,
                                                        MLT, DII, DII);
  }

  // 6) abc = xc @ W_x^T : M=4096, N=32, K=1536 (needs N guard)
  {
    dim3 g(1, (MLT + 63) / 64);
    gemm_bf16_wmma_kernel<true><<<g, 256, 0, stream>>>(xcc, Wxt, abc,
                                                       MLT, 2 * DSS, DII);
  }

  // 7) selective scan (softplus + gating fused) -> ygbf (bf16)
  {
    dim3 g(DII / 256, BB);
    scan_kernel<<<g, 256, 0, stream>>>(abc, dtpre, b_dt, xcc, xz, ygbf);
  }

  // 8) out = y_gated @ W_out^T : M=4096, N=768, K=1536
  {
    dim3 g((DD + 255) / 256, (MLT + 63) / 64);
    gemm_bf16_wmma_kernel<false><<<g, 256, 0, stream>>>(ygbf, Woutt, out,
                                                        MLT, DD, DII);
  }
}
